// PureOrthoPhasor_9251359556066
// MI455X (gfx1250) — matrix operand order repaired and checked by reading, not compiled, not run on hardware
//
#include <hip/hip_runtime.h>
#include <math.h>

// Problem constants (match reference)
#define BATCH 4
#define LEN   4096
#define DIM   1024
#define NROWS (BATCH * LEN)          // 16384 GEMM rows
#define NCH   32                     // scan chunks along L
#define CHL   (LEN / NCH)            // 128 elements per chunk
#define INV_SQRT_D 0.03125f          // 1/sqrt(1024)

typedef __attribute__((ext_vector_type(16))) __bf16 v16bf;
typedef __attribute__((ext_vector_type(8)))  float  v8f;

// ---------------------------------------------------------------------------
// fp32 -> bf16 (round-to-nearest-even), stored as u16
// ---------------------------------------------------------------------------
__device__ __forceinline__ unsigned short f2bf(float f) {
    unsigned int u = __float_as_uint(f);
    unsigned int r = (u + 0x7fffu + ((u >> 16) & 1u)) >> 16;
    return (unsigned short)r;
}

__global__ __launch_bounds__(256) void cvt_f32_bf16(const float* __restrict__ in,
                                                    unsigned short* __restrict__ out,
                                                    int n) {
    int i = blockIdx.x * blockDim.x + threadIdx.x;
    if (i < n) out[i] = f2bf(in[i]);
}

// ---------------------------------------------------------------------------
// WMMA fragment loads (bf16, 16x16x32). K is contiguous in memory for both
// A ([M,K] row-major) and B ([N,K] row-major, since einsum contracts d).
// A 16-bit layout: lane m=lane&15; half=lane>>4; VGPR0-3: K=half*8+[0..7],
// VGPR4-7: K=16+half*8+[0..7]  -> two b128 loads.
// B 32x16 layout: lane holds column N=lane&15; same K order as A.
// ---------------------------------------------------------------------------
__device__ __forceinline__ v16bf load_frag_A(const unsigned short* rowk, int half) {
    union { int4 q[2]; v16bf v; } f;
    f.q[0] = *reinterpret_cast<const int4*>(rowk + half * 8);
    f.q[1] = *reinterpret_cast<const int4*>(rowk + 16 + half * 8);
    return f.v;
}

__device__ __forceinline__ v16bf load_frag_B(const unsigned short* colk, int half) {
    union { int4 q[2]; v16bf v; } f;
    f.q[0] = *reinterpret_cast<const int4*>(colk + half * 8);
    f.q[1] = *reinterpret_cast<const int4*>(colk + 16 + half * 8);
    return f.v;
}

__device__ __forceinline__ v8f wmma_bf16(v16bf a, v16bf b, v8f c) {
    return __builtin_amdgcn_wmma_f32_16x16x32_bf16(
        /*neg_a=*/false, a, /*neg_b=*/false, b,
        /*c_mod=*/(short)0, c, /*reuse_a=*/false, /*reuse_b=*/false);
}

// ---------------------------------------------------------------------------
// C[M,N] = A[M,K] * B[N,K]^T + bias[N] (+ residual[M,N])
// Block = 128 threads = 4 waves, each wave computes a 32x32 tile (2x2 frags).
// Grid: (N/64, M/64).
// ---------------------------------------------------------------------------
__global__ __launch_bounds__(128) void wmma_gemm_bf16(
    const unsigned short* __restrict__ A,
    const unsigned short* __restrict__ Bm,
    const float* __restrict__ bias,
    const float* __restrict__ residual,   // may be nullptr
    float* __restrict__ C,
    int M, int N, int K)
{
    const int lane = threadIdx.x & 31;
    const int wave = threadIdx.x >> 5;
    const int wm = wave >> 1;
    const int wn = wave & 1;
    const int M0 = blockIdx.y * 64 + wm * 32;
    const int N0 = blockIdx.x * 64 + wn * 32;
    const int m    = lane & 15;
    const int half = lane >> 4;

    const unsigned short* arow0 = A  + (size_t)(M0 +      m) * K;
    const unsigned short* arow1 = A  + (size_t)(M0 + 16 + m) * K;
    const unsigned short* brow0 = Bm + (size_t)(N0 +      m) * K;
    const unsigned short* brow1 = Bm + (size_t)(N0 + 16 + m) * K;

    v8f acc00 = {}; v8f acc01 = {}; v8f acc10 = {}; v8f acc11 = {};

    for (int k0 = 0; k0 < K; k0 += 32) {
        v16bf a0 = load_frag_A(arow0 + k0, half);
        v16bf a1 = load_frag_A(arow1 + k0, half);
        v16bf b0 = load_frag_B(brow0 + k0, half);
        v16bf b1 = load_frag_B(brow1 + k0, half);
        acc00 = wmma_bf16(a0, b0, acc00);
        acc01 = wmma_bf16(a0, b1, acc01);
        acc10 = wmma_bf16(a1, b0, acc10);
        acc11 = wmma_bf16(a1, b1, acc11);
    }

    // C/D layout: VGPR r -> M = half*8 + r, N = lane&15
    #pragma unroll
    for (int mi = 0; mi < 2; ++mi) {
        #pragma unroll
        for (int ni = 0; ni < 2; ++ni) {
            v8f acc = (mi == 0) ? (ni == 0 ? acc00 : acc01)
                                : (ni == 0 ? acc10 : acc11);
            const int col = N0 + ni * 16 + (lane & 15);
            #pragma unroll
            for (int r = 0; r < 8; ++r) {
                const int row = M0 + mi * 16 + half * 8 + r;
                float v = acc[r] + bias[col];
                size_t idx = (size_t)row * N + col;
                if (residual) v += residual[idx];
                C[idx] = v;
            }
        }
    }
}

// ---------------------------------------------------------------------------
// Scan pass 1: per-(b, chunk, d) partial sums of value*cos / value*sin.
// Threads coalesce across d. Total threads = BATCH*NCH*DIM = 131072.
// ---------------------------------------------------------------------------
__global__ __launch_bounds__(256) void scan_partials(
    const float* __restrict__ value,
    const float* __restrict__ phases,      // [LEN, DIM]
    float* __restrict__ partR,             // [BATCH, NCH, DIM]
    float* __restrict__ partI)
{
    int idx = blockIdx.x * blockDim.x + threadIdx.x;
    if (idx >= BATCH * NCH * DIM) return;
    const int d = idx % DIM;
    const int c = (idx / DIM) % NCH;
    const int b = idx / (DIM * NCH);

    float sr = 0.f, si = 0.f;
    const int l0 = c * CHL;
    for (int l = l0; l < l0 + CHL; ++l) {
        float p = phases[(size_t)l * DIM + d];
        float cp = __cosf(p), sp = __sinf(p);
        float v = value[((size_t)b * LEN + l) * DIM + d];
        sr += v * cp;
        si += v * sp;
    }
    partR[((size_t)b * NCH + c) * DIM + d] = sr;
    partI[((size_t)b * NCH + c) * DIM + d] = si;
}

// ---------------------------------------------------------------------------
// Scan pass 2: exclusive prefix over chunk partials, then in-chunk running
// sum; emit retrieved = (mem_r*cos + mem_i*sin) / sqrt(D).
// ---------------------------------------------------------------------------
__global__ __launch_bounds__(256) void scan_apply(
    const float* __restrict__ value,
    const float* __restrict__ phases,
    const float* __restrict__ partR,
    const float* __restrict__ partI,
    float* __restrict__ retrieved)         // [BATCH, LEN, DIM]
{
    int idx = blockIdx.x * blockDim.x + threadIdx.x;
    if (idx >= BATCH * NCH * DIM) return;
    const int d = idx % DIM;
    const int c = (idx / DIM) % NCH;
    const int b = idx / (DIM * NCH);

    float mr = 0.f, mi = 0.f;
    for (int cc = 0; cc < c; ++cc) {
        mr += partR[((size_t)b * NCH + cc) * DIM + d];
        mi += partI[((size_t)b * NCH + cc) * DIM + d];
    }

    const int l0 = c * CHL;
    for (int l = l0; l < l0 + CHL; ++l) {
        float p = phases[(size_t)l * DIM + d];
        float cp = __cosf(p), sp = __sinf(p);
        size_t vidx = ((size_t)b * LEN + l) * DIM + d;
        float v = value[vidx];
        mr += v * cp;
        mi += v * sp;
        retrieved[vidx] = (mr * cp + mi * sp) * INV_SQRT_D;
    }
}

// ---------------------------------------------------------------------------
// LayerNorm over D=1024 per row, output bf16 for the second WMMA GEMM.
// One 256-thread block per row; LDS tree reduction.
// ---------------------------------------------------------------------------
__global__ __launch_bounds__(256) void ln_to_bf16(
    const float* __restrict__ X,           // [NROWS, DIM]
    const float* __restrict__ gamma,
    const float* __restrict__ beta,
    unsigned short* __restrict__ Y)        // [NROWS, DIM] bf16
{
    const size_t row = blockIdx.x;
    const float* xr = X + row * DIM;
    const int tid = threadIdx.x;

    float s = 0.f, q = 0.f;
    for (int i = tid; i < DIM; i += 256) {
        float v = xr[i];
        s += v;
        q += v * v;
    }
    __shared__ float shs[256];
    __shared__ float shq[256];
    shs[tid] = s; shq[tid] = q;
    __syncthreads();
    for (int off = 128; off > 0; off >>= 1) {
        if (tid < off) { shs[tid] += shs[tid + off]; shq[tid] += shq[tid + off]; }
        __syncthreads();
    }
    const float mu  = shs[0] * (1.0f / DIM);
    const float var = shq[0] * (1.0f / DIM) - mu * mu;
    const float inv = rsqrtf(var + 1e-5f);

    for (int i = tid; i < DIM; i += 256) {
        float y = (xr[i] - mu) * inv * gamma[i] + beta[i];
        Y[row * DIM + i] = f2bf(y);
    }
}

// ---------------------------------------------------------------------------
// Host-side launch
// ---------------------------------------------------------------------------
extern "C" void kernel_launch(void* const* d_in, const int* in_sizes, int n_in,
                              void* d_out, int out_size, void* d_ws, size_t ws_size,
                              hipStream_t stream) {
    (void)in_sizes; (void)n_in; (void)out_size; (void)ws_size;

    const float* x       = (const float*)d_in[0];   // [B,L,D]
    const float* phases  = (const float*)d_in[1];   // [L,D]
    const float* W_v     = (const float*)d_in[2];   // [D,D]
    const float* b_v     = (const float*)d_in[3];   // [D]
    const float* ln_g    = (const float*)d_in[4];   // [D]
    const float* ln_b    = (const float*)d_in[5];   // [D]
    const float* W_o     = (const float*)d_in[6];   // [D,D]
    const float* b_o     = (const float*)d_in[7];   // [D]
    float* out = (float*)d_out;                     // [B,L,D]

    // Workspace layout (256B aligned slabs)
    char* ws = (char*)d_ws;
    size_t off = 0;
    auto alloc = [&](size_t bytes) {
        char* p = ws + off;
        off = (off + bytes + 255) & ~(size_t)255;
        return p;
    };
    unsigned short* xb   = (unsigned short*)alloc((size_t)NROWS * DIM * 2);  // bf16 x
    unsigned short* wvb  = (unsigned short*)alloc((size_t)DIM * DIM * 2);    // bf16 W_v
    unsigned short* wob  = (unsigned short*)alloc((size_t)DIM * DIM * 2);    // bf16 W_o
    float* value         = (float*)alloc((size_t)NROWS * DIM * 4);
    float* partR         = (float*)alloc((size_t)BATCH * NCH * DIM * 4);
    float* partI         = (float*)alloc((size_t)BATCH * NCH * DIM * 4);
    float* retrieved     = (float*)alloc((size_t)NROWS * DIM * 4);
    unsigned short* lnb16= (unsigned short*)alloc((size_t)NROWS * DIM * 2);  // bf16 LN(retrieved)

    // 1) bf16 conversions
    {
        int n = NROWS * DIM;
        cvt_f32_bf16<<<(n + 255) / 256, 256, 0, stream>>>(x, xb, n);
        n = DIM * DIM;
        cvt_f32_bf16<<<(n + 255) / 256, 256, 0, stream>>>(W_v, wvb, n);
        cvt_f32_bf16<<<(n + 255) / 256, 256, 0, stream>>>(W_o, wob, n);
    }

    // 2) value = x @ W_v^T + b_v
    {
        dim3 grid(DIM / 64, NROWS / 64);
        wmma_gemm_bf16<<<grid, 128, 0, stream>>>(xb, wvb, b_v, nullptr, value,
                                                 NROWS, DIM, DIM);
    }

    // 3-4) phasor cumulative scan -> retrieved
    {
        int n = BATCH * NCH * DIM;
        scan_partials<<<(n + 255) / 256, 256, 0, stream>>>(value, phases, partR, partI);
        scan_apply<<<(n + 255) / 256, 256, 0, stream>>>(value, phases, partR, partI,
                                                        retrieved);
    }

    // 5) LayerNorm -> bf16
    ln_to_bf16<<<NROWS, 256, 0, stream>>>(retrieved, ln_g, ln_b, lnb16);

    // 6) out = LN @ W_o^T + b_o + x   (residual fused)
    {
        dim3 grid(DIM / 64, NROWS / 64);
        wmma_gemm_bf16<<<grid, 128, 0, stream>>>(lnb16, wob, b_o, x, out,
                                                 NROWS, DIM, DIM);
    }
}
